// MultiHeadAttention_16879221473666
// MI455X (gfx1250) — matrix operand order repaired and checked by reading, not compiled, not run on hardware
//
#include <hip/hip_runtime.h>
#include <hip/hip_bf16.h>
#include <stdint.h>

// ---------------------------------------------------------------------------
// MHA forward for MI455X (gfx1250, wave32): all GEMMs via v_wmma_f32_16x16x32_f16
// ---------------------------------------------------------------------------
typedef __attribute__((ext_vector_type(16))) _Float16 v16h;
typedef __attribute__((ext_vector_type(8)))  _Float16 v8h;
typedef __attribute__((ext_vector_type(8)))  float    v8f;

#define B_   8
#define S_   1024
#define H_   16
#define DK_  128
#define DM_  512
#define HD_  2048   // H_*DK_

__device__ __forceinline__ v8f wmma16(v16h a, v16h b, v8f c) {
  // D = A(16x32 f16) * B(32x16 f16) + C(16x16 f32)
  return __builtin_amdgcn_wmma_f32_16x16x32_f16(
      /*neg_a=*/false, a, /*neg_b=*/false, b,
      /*c_mod=*/(short)0, c, /*reuse_a=*/false, /*reuse_b=*/false);
}

// A-fragment (16x32, f16 source, row-major, ld elems).
// Lane r<16 holds K=0..7,16..23 ; lane r+16 holds K=8..15,24..31.
__device__ __forceinline__ v16h fragA_f16(const _Float16* __restrict__ base, int ld) {
  const int lane = threadIdx.x & 31;
  const int r  = lane & 15;
  const int kh = (lane >> 4) << 3;               // 0 or 8
  const _Float16* p = base + (size_t)r * ld + kh;
  v8h lo = *(const v8h*)(p);                     // K = kh .. kh+7
  v8h hi = *(const v8h*)(p + 16);                // K = 16+kh .. 16+kh+7
  return __builtin_shufflevector(lo, hi, 0,1,2,3,4,5,6,7,8,9,10,11,12,13,14,15);
}

// A-fragment from an f32 source (convert in registers).
__device__ __forceinline__ v16h fragA_f32(const float* __restrict__ base, int ld) {
  const int lane = threadIdx.x & 31;
  const int r  = lane & 15;
  const int kh = (lane >> 4) << 3;
  const float* p = base + (size_t)r * ld + kh;
  v8f lo = *(const v8f*)(p);
  v8f hi = *(const v8f*)(p + 16);
  v8h l = __builtin_convertvector(lo, v8h);
  v8h h = __builtin_convertvector(hi, v8h);
  return __builtin_shufflevector(l, h, 0,1,2,3,4,5,6,7,8,9,10,11,12,13,14,15);
}

// B-fragment (32x16): source is B^T [N][K] row-major. Lane c<16 holds
// K=0..7,16..23 ; lane c+16 holds K=8..15,24..31 (two 16-byte loads).
__device__ __forceinline__ v16h fragB_f16(const _Float16* __restrict__ bt, int ld) {
  const int lane = threadIdx.x & 31;
  const int c  = lane & 15;
  const int kh = (lane >> 4) << 3;               // 0 or 8
  const _Float16* p = bt + (size_t)c * ld + kh;
  v8h lo = *(const v8h*)(p);
  v8h hi = *(const v8h*)(p + 16);
  return __builtin_shufflevector(lo, hi, 0,1,2,3,4,5,6,7,8,9,10,11,12,13,14,15);
}

// --------------------------- kernel 0: weight transpose + f16 convert -------
__global__ void k_wt_cvt(const float* __restrict__ W, _Float16* __restrict__ Wt,
                         int K, int N) {
  int idx = blockIdx.x * blockDim.x + threadIdx.x;
  if (idx >= K * N) return;
  int n = idx / K, k = idx - n * K;              // Wt[n][k] = W[k][n]
  Wt[(size_t)n * K + k] = (_Float16)W[(size_t)k * N + n];
}

// --------------------------- kernel 1: input projections --------------------
// C[8192,2048] = X[8192,512] @ W[512,2048]; one 16x16 tile per wave.
// vmode=0: out = [B,H,S,128] f16 (Q,K).  vmode=1: out = [B,H,128,S] f16 (V^T).
__global__ void k_proj(const float* __restrict__ X, const _Float16* __restrict__ Wt,
                       _Float16* __restrict__ out, int vmode) {
  const int wave = blockIdx.x * (blockDim.x >> 5) + (threadIdx.x >> 5);
  const int m0 = (wave >> 7) << 4;               // 512 row tiles
  const int n0 = (wave & 127) << 4;              // 128 col tiles
  v8f acc = {};
  for (int k = 0; k < DM_; k += 32) {
    v16h a = fragA_f32(X + (size_t)m0 * DM_ + k, DM_);
    v16h b = fragB_f16(Wt + (size_t)n0 * DM_ + k, DM_);
    acc = wmma16(a, b, acc);
  }
  const int lane = threadIdx.x & 31;
  const int n = n0 + (lane & 15);
  const int h = n >> 7, d = n & 127;
#pragma unroll
  for (int i = 0; i < 8; ++i) {
    int m = m0 + ((lane >> 4) << 3) + i;
    int bb = m >> 10, s = m & 1023;
    size_t bh = (size_t)bb * H_ + h;
    if (vmode) out[(bh * DK_ + d) * S_ + s] = (_Float16)acc[i];
    else       out[(bh * S_  + s) * DK_ + d] = (_Float16)acc[i];
  }
}

// ------------- kernel 2: scores = QK^T/sqrt(d), mask, softmax ---------------
// One block = (b,h, 16 q-rows). 8 waves x 8 col-tiles => 16x1024 scores in regs.
__global__ void k_scores(const _Float16* __restrict__ Qh, const _Float16* __restrict__ Kh,
                         const uint8_t* __restrict__ mask, float* __restrict__ attnO) {
  const int qt = blockIdx.x & 63;
  const int bh = blockIdx.x >> 6;
  const int b  = bh >> 4;
  const int q0 = qt << 4;
  const int w    = threadIdx.x >> 5;
  const int lane = threadIdx.x & 31;
  const int rb = (lane >> 4) << 3;               // row base within tile
  const int cl = lane & 15;

  const _Float16* Qb = Qh + ((size_t)bh * S_ + q0) * DK_;
  const _Float16* Kb = Kh + (size_t)bh * S_ * DK_;

  v16h aq[4];
#pragma unroll
  for (int ks = 0; ks < 4; ++ks) aq[ks] = fragA_f16(Qb + ks * 32, DK_);

  v8f acc[8];
#pragma unroll
  for (int t = 0; t < 8; ++t) acc[t] = (v8f){};
#pragma unroll
  for (int ks = 0; ks < 4; ++ks)
#pragma unroll
    for (int t = 0; t < 8; ++t) {
      v16h bf = fragB_f16(Kb + (size_t)(w * 128 + t * 16) * DK_ + ks * 32, DK_);
      acc[t] = wmma16(aq[ks], bf, acc[t]);
    }

  // mask + scale, per-lane row maxima over this wave's 128 columns
  const float scale = 0.088388347648318447f;     // 1/sqrt(128)
  float pmax[8];
#pragma unroll
  for (int i = 0; i < 8; ++i) pmax[i] = -3.0e38f;
#pragma unroll
  for (int t = 0; t < 8; ++t)
#pragma unroll
    for (int i = 0; i < 8; ++i) {
      int q = q0 + rb + i, kp = w * 128 + t * 16 + cl;
      float s = mask[((size_t)b * S_ + q) * S_ + kp] ? -1.0e9f : acc[t][i] * scale;
      acc[t][i] = s;
      pmax[i] = fmaxf(pmax[i], s);
    }
#pragma unroll
  for (int i = 0; i < 8; ++i)
#pragma unroll
    for (int off = 1; off < 16; off <<= 1)
      pmax[i] = fmaxf(pmax[i], __shfl_xor(pmax[i], off, 32));

  __shared__ float red[8][16];
  if (cl == 0)
#pragma unroll
    for (int i = 0; i < 8; ++i) red[w][rb + i] = pmax[i];
  __syncthreads();
  float gmax[8];
#pragma unroll
  for (int i = 0; i < 8; ++i) {
    float m = red[0][rb + i];
#pragma unroll
    for (int ww = 1; ww < 8; ++ww) m = fmaxf(m, red[ww][rb + i]);
    gmax[i] = m;
  }
  __syncthreads();

  float psum[8];
#pragma unroll
  for (int i = 0; i < 8; ++i) psum[i] = 0.f;
#pragma unroll
  for (int t = 0; t < 8; ++t)
#pragma unroll
    for (int i = 0; i < 8; ++i) {
      float p = __expf(acc[t][i] - gmax[i]);
      acc[t][i] = p;
      psum[i] += p;
    }
#pragma unroll
  for (int i = 0; i < 8; ++i)
#pragma unroll
    for (int off = 1; off < 16; off <<= 1)
      psum[i] += __shfl_xor(psum[i], off, 32);
  if (cl == 0)
#pragma unroll
    for (int i = 0; i < 8; ++i) red[w][rb + i] = psum[i];
  __syncthreads();
  float ginv[8];
#pragma unroll
  for (int i = 0; i < 8; ++i) {
    float s = 0.f;
#pragma unroll
    for (int ww = 0; ww < 8; ++ww) s += red[ww][rb + i];
    ginv[i] = 1.0f / s;
  }

  float* aout = attnO + ((size_t)bh * S_ + q0) * S_;
#pragma unroll
  for (int t = 0; t < 8; ++t)
#pragma unroll
    for (int i = 0; i < 8; ++i)
      aout[(size_t)(rb + i) * S_ + w * 128 + t * 16 + cl] = acc[t][i] * ginv[i];
}

// --------------------- kernel 3: context = attn @ V -------------------------
__global__ void k_ctx(const float* __restrict__ attnO, const _Float16* __restrict__ Vt,
                      _Float16* __restrict__ ctx) {
  const int qt = blockIdx.x & 63;
  const int bh = blockIdx.x >> 6;
  const int b = bh >> 4, h = bh & 15;
  const int q0 = qt << 4;
  const int w    = threadIdx.x >> 5;
  const int lane = threadIdx.x & 31;
  const float*    Ab = attnO + ((size_t)bh * S_ + q0) * S_;
  const _Float16* Vb = Vt + ((size_t)bh * DK_ + w * 16) * S_;   // V^T rows = d
  v8f acc = {};
  for (int k = 0; k < S_; k += 32) {
    v16h a  = fragA_f32(Ab + k, S_);
    v16h bf = fragB_f16(Vb + k, S_);
    acc = wmma16(a, bf, acc);
  }
  const int n = h * DK_ + w * 16 + (lane & 15);
#pragma unroll
  for (int i = 0; i < 8; ++i) {
    int s = q0 + ((lane >> 4) << 3) + i;
    ctx[((size_t)b * S_ + s) * HD_ + n] = (_Float16)acc[i];
  }
}

// --------- kernel 4: out = LN(ctx @ W_fc + input_Q) -------------------------
__global__ void k_fc_ln(const _Float16* __restrict__ ctx, const _Float16* __restrict__ Wfct,
                        const float* __restrict__ resid, float* __restrict__ out) {
  const int m0 = blockIdx.x << 4;                // 16 rows per block
  const int w    = threadIdx.x >> 5;
  const int lane = threadIdx.x & 31;
  const int rb = (lane >> 4) << 3;
  const int cl = lane & 15;

  v8f acc[4];
#pragma unroll
  for (int j = 0; j < 4; ++j) acc[j] = (v8f){};
  for (int k = 0; k < HD_; k += 32) {
    v16h a = fragA_f16(ctx + (size_t)m0 * HD_ + k, HD_);
#pragma unroll
    for (int j = 0; j < 4; ++j) {
      v16h bf = fragB_f16(Wfct + (size_t)(w * 64 + j * 16) * HD_ + k, HD_);
      acc[j] = wmma16(a, bf, acc[j]);
    }
  }
  // residual + row sums for LayerNorm
  float p1[8], p2[8];
#pragma unroll
  for (int i = 0; i < 8; ++i) { p1[i] = 0.f; p2[i] = 0.f; }
#pragma unroll
  for (int j = 0; j < 4; ++j)
#pragma unroll
    for (int i = 0; i < 8; ++i) {
      int m = m0 + rb + i, n = w * 64 + j * 16 + cl;
      float y = acc[j][i] + resid[(size_t)m * DM_ + n];
      acc[j][i] = y;
      p1[i] += y; p2[i] += y * y;
    }
#pragma unroll
  for (int i = 0; i < 8; ++i)
#pragma unroll
    for (int off = 1; off < 16; off <<= 1) {
      p1[i] += __shfl_xor(p1[i], off, 32);
      p2[i] += __shfl_xor(p2[i], off, 32);
    }
  __shared__ float s1[8][16], s2[8][16];
  if (cl == 0)
#pragma unroll
    for (int i = 0; i < 8; ++i) { s1[w][rb + i] = p1[i]; s2[w][rb + i] = p2[i]; }
  __syncthreads();
  float mean[8], rinv[8];
#pragma unroll
  for (int i = 0; i < 8; ++i) {
    float t1 = 0.f, t2 = 0.f;
#pragma unroll
    for (int ww = 0; ww < 8; ++ww) { t1 += s1[ww][rb + i]; t2 += s2[ww][rb + i]; }
    float mu = t1 * (1.0f / DM_);
    float var = t2 * (1.0f / DM_) - mu * mu;
    mean[i] = mu;
    rinv[i] = rsqrtf(var + 1e-5f);
  }
#pragma unroll
  for (int j = 0; j < 4; ++j)
#pragma unroll
    for (int i = 0; i < 8; ++i) {
      int m = m0 + rb + i, n = w * 64 + j * 16 + cl;
      out[(size_t)m * DM_ + n] = (acc[j][i] - mean[i]) * rinv[i];
    }
}

// ---------------------------------------------------------------------------
extern "C" void kernel_launch(void* const* d_in, const int* in_sizes, int n_in,
                              void* d_out, int out_size, void* d_ws, size_t ws_size,
                              hipStream_t stream) {
  const float*   inQ  = (const float*)d_in[0];
  const float*   inK  = (const float*)d_in[1];
  const float*   inV  = (const float*)d_in[2];
  const uint8_t* mask = (const uint8_t*)d_in[3];
  const float*   WQ   = (const float*)d_in[4];
  const float*   WK   = (const float*)d_in[5];
  const float*   WV   = (const float*)d_in[6];
  const float*   Wfc  = (const float*)d_in[7];

  float* out   = (float*)d_out;
  float* attnO = out + (size_t)B_ * S_ * DM_;    // attn matrix lives in d_out

  // workspace layout (f16)
  _Float16* p   = (_Float16*)d_ws;
  _Float16* WtQ  = p;  p += (size_t)DM_ * HD_;   // [2048][512]
  _Float16* WtK  = p;  p += (size_t)DM_ * HD_;
  _Float16* WtV  = p;  p += (size_t)DM_ * HD_;
  _Float16* Wfct = p;  p += (size_t)HD_ * DM_;   // [512][2048]
  _Float16* Qh   = p;  p += (size_t)B_ * H_ * S_ * DK_;  // [B,H,S,128]
  _Float16* Kh   = p;  p += (size_t)B_ * H_ * S_ * DK_;
  _Float16* Vt   = p;  p += (size_t)B_ * H_ * DK_ * S_;  // [B,H,128,S]
  _Float16* ctx  = p;                                    // [B,S,2048]

  const int wcnt = DM_ * HD_;                    // 1,048,576 elements per weight
  k_wt_cvt<<<(wcnt + 255) / 256, 256, 0, stream>>>(WQ,  WtQ,  DM_, HD_);
  k_wt_cvt<<<(wcnt + 255) / 256, 256, 0, stream>>>(WK,  WtK,  DM_, HD_);
  k_wt_cvt<<<(wcnt + 255) / 256, 256, 0, stream>>>(WV,  WtV,  DM_, HD_);
  k_wt_cvt<<<(wcnt + 255) / 256, 256, 0, stream>>>(Wfc, Wfct, HD_, DM_);

  k_proj<<<8192, 256, 0, stream>>>(inQ, WtQ, Qh, 0);
  k_proj<<<8192, 256, 0, stream>>>(inK, WtK, Kh, 0);
  k_proj<<<8192, 256, 0, stream>>>(inV, WtV, Vt, 1);

  k_scores<<<B_ * H_ * (S_ / 16), 256, 0, stream>>>(Qh, Kh, mask, attnO);
  k_ctx   <<<B_ * H_ * (S_ / 16), 256, 0, stream>>>(attnO, Vt, ctx);
  k_fc_ln <<<(B_ * S_) / 16, 256, 0, stream>>>(ctx, Wfct, inQ, out);
}
